// BiLSTM_73787538145449
// MI455X (gfx1250) — matrix operand order repaired and checked by reading, not compiled, not run on hardware
//
#include <hip/hip_runtime.h>
#include <math.h>

// ---------------------------------------------------------------------------
// BiLSTM (B=64, T=512, D_IN=512, H=1024) + final linear, for gfx1250.
//
//  * f16 data / f32-accum WMMA (v_wmma_f32_16x16x32_f16).
//  * Persistent recurrent kernel: 64 WGs/direction, 16 hidden units each.
//    - W_hh slice resident in LDS (async-copied once).
//    - h_{t-1} async-copied global->LDS each step, overlapped with the
//      x_t @ W_ih^T WMMA loop (global_load_async_to_lds_b128 + asynccnt).
//    - pointwise LSTM in f32, c in registers, h ping-pong in global.
//    - per-direction atomic grid barrier + agent fences each step.
//  * Final [64,2048] @ W_lin^T with the same WMMA tiling.
// ---------------------------------------------------------------------------

#define BB    64
#define TT    512
#define DIN   512
#define HH    1024
#define G4    4096      // 4*H
#define H2    2048      // 2*H
#define DOUT  512
#define NWG_DIR 64      // workgroups per direction
#define JH    16        // hidden units per workgroup
#define SWPITCH 1032    // padded LDS row pitch in halves (2064B, 16B aligned)

typedef __attribute__((ext_vector_type(16))) _Float16 v16h;
typedef __attribute__((ext_vector_type(8)))  float    v8f;

union V16 { v16h v; uint4 q[2]; };

// A fragment (16x32 f16, row-major source, row stride rstride halves).
// lane = lr + 16*lh ; lanes 0-15: K chunks [0,8)+[16,24); lanes 16-31: +8.
__device__ __forceinline__ v16h load_a16(const _Float16* base, size_t rstride,
                                         int m, int kbase, int lh) {
  const _Float16* p = base + (size_t)m * rstride + (size_t)(kbase + lh * 8);
  V16 u;
  u.q[0] = *(const uint4*)(p);
  u.q[1] = *(const uint4*)(p + 16);
  return u.v;
}

// B fragment (32x16 f16) from a row-major weight row (B[k][n] = W[n][k]):
// lane lr holds N=lr; lanes 0-15: K chunks [0,8)+[16,24); lanes 16-31: +8.
__device__ __forceinline__ v16h load_b16(const _Float16* rowp, int kbase, int lh) {
  const _Float16* p = rowp + kbase + lh * 8;
  V16 u;
  u.q[0] = *(const uint4*)(p);
  u.q[1] = *(const uint4*)(p + 16);
  return u.v;
}

// Async 16B global -> LDS copy (CDNA5 async path, tracked by ASYNCcnt).
// dst: LDS byte offset (low 32 bits of a shared-aperture generic pointer).
__device__ __forceinline__ void async_copy16(unsigned lds_off, const void* gsrc) {
  unsigned long long ga = (unsigned long long)(uintptr_t)gsrc;
  asm volatile("global_load_async_to_lds_b128 %0, %1, off"
               :: "v"(lds_off), "v"(ga)
               : "memory");
}
__device__ __forceinline__ void wait_async0() {
  asm volatile("s_wait_asynccnt 0x0" ::: "memory");
}

__device__ __forceinline__ float fast_sigmoid(float x) {
  return 1.0f / (1.0f + __expf(-x));
}
__device__ __forceinline__ float fast_tanh(float x) {
  return 1.0f - 2.0f / (__expf(2.0f * x) + 1.0f);
}

// ---------------------------------------------------------------------------
// Persistent bidirectional LSTM kernel. Grid = 128 blocks x 256 threads.
// blocks 0..63 = forward direction, 64..127 = backward.
// ---------------------------------------------------------------------------
__global__ __launch_bounds__(256, 1) void lstm_kernel(
    const _Float16* __restrict__ xf16,   // [B][T][DIN] f16
    const _Float16* __restrict__ whh,    // [2][4096][1024] f16
    const _Float16* __restrict__ wih,    // [2][4096][512]  f16
    const float*    __restrict__ bias,   // [2][4096] f32 (b_ih + b_hh)
    _Float16*       __restrict__ hbuf,   // [2][2][B][H] f16 ping-pong
    int*            __restrict__ bar)    // [4] ints: cnt[2], gen[2]
{
  __shared__ __align__(16) _Float16 sW[64 * SWPITCH]; // W_hh slice (129KB)
  __shared__ __align__(16) _Float16 sH[64 * SWPITCH]; // h_{t-1} stage (129KB)
  __shared__ __align__(16) float    sG[64 * 64];      // gate tile (16KB)

  const int tid = threadIdx.x;
  const int wg  = blockIdx.x;
  const int dir = wg >> 6;         // 0 = fwd, 1 = bwd
  const int wgd = wg & 63;         // workgroup within direction
  const int j0  = wgd * JH;        // first hidden unit owned

  const int w   = tid >> 5;        // wave 0..7
  const int ln  = tid & 31;
  const int lh  = ln >> 4;         // lane half
  const int lr  = ln & 15;
  const int m0  = (w & 3) * 16;    // M tile base (batch rows)
  const int nt0 = (w >> 2) * 2;    // first of two N tiles (gate index)

  // ---- async-stage W_hh slice into LDS, rows in tile order n = g*16+jj ----
  const _Float16* whhd = whh + (size_t)dir * G4 * HH;
  for (int idx = tid; idx < 64 * 128; idx += 256) {
    const int r  = idx >> 7;            // LDS row 0..63
    const int ck = idx & 127;           // 16B chunk within row
    const int grow = (r >> 4) * HH + j0 + (r & 15);  // global gate row
    async_copy16((unsigned)(uintptr_t)&sW[r * SWPITCH + ck * 8],
                 whhd + (size_t)grow * HH + ck * 8);
  }

  // ---- per-thread pointwise state: thread owns (jj, 4 batch rows) ----
  const int jj = tid & 15;
  const int mb = (tid >> 4) * 4;
  float cc[4] = {0.0f, 0.0f, 0.0f, 0.0f};
  const float bi = bias[dir * G4 + 0 * HH + j0 + jj];
  const float bf = bias[dir * G4 + 1 * HH + j0 + jj];
  const float bg = bias[dir * G4 + 2 * HH + j0 + jj];
  const float bo = bias[dir * G4 + 3 * HH + j0 + jj];

  // ---- constant B-row pointers ----
  const _Float16* wihd = wih + (size_t)dir * G4 * DIN;
  const _Float16* bih0 = wihd + (size_t)((nt0 + 0) * HH + j0 + lr) * DIN;
  const _Float16* bih1 = wihd + (size_t)((nt0 + 1) * HH + j0 + lr) * DIN;
  const _Float16* sW0  = &sW[((nt0 + 0) * 16 + lr) * SWPITCH];
  const _Float16* sW1  = &sW[((nt0 + 1) * 16 + lr) * SWPITCH];

  wait_async0();
  __syncthreads();  // sW ready

  int* cnt = bar + dir;
  int* gen = bar + 2 + dir;

  for (int t = 0; t < TT; ++t) {
    const int tt = dir ? (TT - 1 - t) : t;
    const _Float16* hr = hbuf + (size_t)(dir * 2 + (t & 1)) * BB * HH;
    _Float16*       hw = hbuf + (size_t)(dir * 2 + ((t + 1) & 1)) * BB * HH;
    const _Float16* xb = xf16 + (size_t)tt * DIN;  // + m*(T*DIN)

    // ---- kick off async h_{t-1} -> LDS copy (overlaps with x-GEMM) ----
    for (int idx = tid; idx < 64 * 128; idx += 256) {
      const int r  = idx >> 7;
      const int ck = idx & 127;
      async_copy16((unsigned)(uintptr_t)&sH[r * SWPITCH + ck * 8],
                   hr + (size_t)r * HH + ck * 8);
    }

    if (t + 1 < TT) {  // prefetch next timestep's x row for this lane's M row
      const int tn = dir ? (TT - 2 - t) : (t + 1);
      __builtin_prefetch(xf16 + ((size_t)(m0 + lr) * TT + tn) * DIN, 0, 1);
    }

    v8f acc0 = {0.f, 0.f, 0.f, 0.f, 0.f, 0.f, 0.f, 0.f};
    v8f acc1 = {0.f, 0.f, 0.f, 0.f, 0.f, 0.f, 0.f, 0.f};

    // ---- x_t @ W_ih^T  (K = 512, A + B streamed from L2) ----
    #pragma unroll 4
    for (int kc = 0; kc < DIN / 32; ++kc) {
      const v16h a  = load_a16(xb, (size_t)TT * DIN, m0 + lr, kc * 32, lh);
      const v16h b0 = load_b16(bih0, kc * 32, lh);
      const v16h b1 = load_b16(bih1, kc * 32, lh);
      acc0 = __builtin_amdgcn_wmma_f32_16x16x32_f16(false, a, false, b0,
                                                    (short)0, acc0, false, false);
      acc1 = __builtin_amdgcn_wmma_f32_16x16x32_f16(false, a, false, b1,
                                                    (short)0, acc1, false, false);
    }

    // ---- wait for staged h, then h_{t-1} @ W_hh^T (K = 1024, all-LDS) ----
    wait_async0();
    __syncthreads();

    #pragma unroll 4
    for (int kc = 0; kc < HH / 32; ++kc) {
      const v16h a  = load_a16(sH, (size_t)SWPITCH, m0 + lr, kc * 32, lh);
      const v16h b0 = load_b16(sW0, kc * 32, lh);
      const v16h b1 = load_b16(sW1, kc * 32, lh);
      acc0 = __builtin_amdgcn_wmma_f32_16x16x32_f16(false, a, false, b0,
                                                    (short)0, acc0, false, false);
      acc1 = __builtin_amdgcn_wmma_f32_16x16x32_f16(false, a, false, b1,
                                                    (short)0, acc1, false, false);
    }

    // ---- D fragments -> LDS gate tile ----
    #pragma unroll
    for (int r = 0; r < 8; ++r) {
      const int m = m0 + lh * 8 + r;          // C/D layout: vgpr r, lane half
      sG[m * 64 + (nt0 + 0) * 16 + lr] = acc0[r];
      sG[m * 64 + (nt0 + 1) * 16 + lr] = acc1[r];
    }
    __syncthreads();

    // ---- pointwise LSTM cell (f32), write h slice f16 ----
    #pragma unroll
    for (int q = 0; q < 4; ++q) {
      const int m = mb + q;
      const float gi = sG[m * 64 +  0 + jj] + bi;
      const float gf = sG[m * 64 + 16 + jj] + bf;
      const float gg = sG[m * 64 + 32 + jj] + bg;
      const float go = sG[m * 64 + 48 + jj] + bo;
      const float ig = fast_sigmoid(gi);
      const float fg = fast_sigmoid(gf);
      const float gz = fast_tanh(gg);
      const float og = fast_sigmoid(go);
      cc[q] = fg * cc[q] + ig * gz;
      const float hv = og * fast_tanh(cc[q]);
      hw[(size_t)m * HH + j0 + jj] = (_Float16)hv;
    }

    // ---- per-direction grid barrier (64 WGs) ----
    __builtin_amdgcn_fence(__ATOMIC_RELEASE, "agent");  // flush h stores
    __syncthreads();
    if (tid == 0) {
      const int g = __hip_atomic_load(gen, __ATOMIC_ACQUIRE,
                                      __HIP_MEMORY_SCOPE_AGENT);
      const int old = __hip_atomic_fetch_add(cnt, 1, __ATOMIC_ACQ_REL,
                                             __HIP_MEMORY_SCOPE_AGENT);
      if (old == NWG_DIR - 1) {
        __hip_atomic_store(cnt, 0, __ATOMIC_RELAXED, __HIP_MEMORY_SCOPE_AGENT);
        __hip_atomic_fetch_add(gen, 1, __ATOMIC_RELEASE,
                               __HIP_MEMORY_SCOPE_AGENT);
      } else {
        while (__hip_atomic_load(gen, __ATOMIC_ACQUIRE,
                                 __HIP_MEMORY_SCOPE_AGENT) == g) {
          __builtin_amdgcn_s_sleep(2);
        }
      }
    }
    __syncthreads();
    // invalidate WGP$ so next step's h loads see other WGPs' stores
    __builtin_amdgcn_fence(__ATOMIC_ACQUIRE, "agent");
  }
}

// ---------------------------------------------------------------------------
// out[64,512] = concat(h_fwd, h_bwd)[64,2048] @ W_lin^T + b_lin
// 128 tiles of 16x16 -> 16 blocks x 8 waves, one tile per wave.
// ---------------------------------------------------------------------------
__global__ __launch_bounds__(256, 1) void linear_kernel(
    const _Float16* __restrict__ hbuf,
    const _Float16* __restrict__ wlin,   // [512][2048] f16
    const float*    __restrict__ blin,   // [512] f32
    float*          __restrict__ out)    // [64][512] f32
{
  const int tid = threadIdx.x;
  const int w  = tid >> 5;
  const int ln = tid & 31;
  const int lh = ln >> 4;
  const int lr = ln & 15;
  const int tile = blockIdx.x * 8 + w;   // 0..127
  const int m0 = (tile & 3) * 16;
  const int n0 = (tile >> 2) * 16;

  const _Float16* hf = hbuf;                          // dir0, buf0 (final)
  const _Float16* hb = hbuf + (size_t)2 * BB * HH;    // dir1, buf0 (final)
  const _Float16* brow = wlin + (size_t)(n0 + lr) * H2;

  v8f acc = {0.f, 0.f, 0.f, 0.f, 0.f, 0.f, 0.f, 0.f};

  #pragma unroll 4
  for (int kc = 0; kc < HH / 32; ++kc) {              // k in [0,1024): h_fwd
    const v16h a = load_a16(hf, (size_t)HH, m0 + lr, kc * 32, lh);
    const v16h b = load_b16(brow, kc * 32, lh);
    acc = __builtin_amdgcn_wmma_f32_16x16x32_f16(false, a, false, b,
                                                 (short)0, acc, false, false);
  }
  #pragma unroll 4
  for (int kc = 0; kc < HH / 32; ++kc) {              // k in [1024,2048): h_bwd
    const v16h a = load_a16(hb, (size_t)HH, m0 + lr, kc * 32, lh);
    const v16h b = load_b16(brow, HH + kc * 32, lh);
    acc = __builtin_amdgcn_wmma_f32_16x16x32_f16(false, a, false, b,
                                                 (short)0, acc, false, false);
  }

  const float bv = blin[n0 + lr];
  #pragma unroll
  for (int r = 0; r < 8; ++r) {
    out[(size_t)(m0 + lh * 8 + r) * DOUT + n0 + lr] = acc[r] + bv;
  }
}

// ---------------------------------------------------------------------------
// Helpers: f32 -> f16 convert, bias sum, zero init.
// ---------------------------------------------------------------------------
__global__ void cvt_kernel(const float* __restrict__ s,
                           _Float16* __restrict__ d, int n) {
  for (int i = blockIdx.x * blockDim.x + threadIdx.x; i < n;
       i += gridDim.x * blockDim.x)
    d[i] = (_Float16)s[i];
}

__global__ void bias_kernel(const float* __restrict__ a,
                            const float* __restrict__ b,
                            float* __restrict__ d, int n) {
  for (int i = blockIdx.x * blockDim.x + threadIdx.x; i < n;
       i += gridDim.x * blockDim.x)
    d[i] = a[i] + b[i];
}

__global__ void zero_kernel(_Float16* __restrict__ d, int n) {
  for (int i = blockIdx.x * blockDim.x + threadIdx.x; i < n;
       i += gridDim.x * blockDim.x)
    d[i] = (_Float16)0.0f;
}

__global__ void zbar_kernel(int* __restrict__ bar) {
  if (threadIdx.x < 4) bar[threadIdx.x] = 0;
}

// ---------------------------------------------------------------------------
extern "C" void kernel_launch(void* const* d_in, const int* in_sizes, int n_in,
                              void* d_out, int out_size, void* d_ws, size_t ws_size,
                              hipStream_t stream) {
  (void)in_sizes; (void)n_in; (void)out_size; (void)ws_size;

  const float* x     = (const float*)d_in[0];
  const float* Wih_f = (const float*)d_in[1];
  const float* Whh_f = (const float*)d_in[2];
  const float* bih_f = (const float*)d_in[3];
  const float* bhh_f = (const float*)d_in[4];
  const float* Wih_b = (const float*)d_in[5];
  const float* Whh_b = (const float*)d_in[6];
  const float* bih_b = (const float*)d_in[7];
  const float* bhh_b = (const float*)d_in[8];
  const float* Wlin  = (const float*)d_in[9];
  const float* blin  = (const float*)d_in[10];
  float* out = (float*)d_out;

  // workspace layout (~62 MB total)
  char* p = (char*)d_ws;
  _Float16* xf16   = (_Float16*)p;  p += (size_t)BB * TT * DIN * 2;   // 33.6MB
  _Float16* whh16  = (_Float16*)p;  p += (size_t)2 * G4 * HH * 2;     // 16.8MB
  _Float16* wih16  = (_Float16*)p;  p += (size_t)2 * G4 * DIN * 2;    //  8.4MB
  _Float16* wlin16 = (_Float16*)p;  p += (size_t)DOUT * H2 * 2;       //  2.1MB
  float*    biasc  = (float*)p;     p += (size_t)2 * G4 * 4;          //  32KB
  _Float16* hbuf   = (_Float16*)p;  p += (size_t)4 * BB * HH * 2;     // 512KB
  int*      bar    = (int*)p;

  cvt_kernel<<<4096, 256, 0, stream>>>(x, xf16, BB * TT * DIN);
  cvt_kernel<<<2048, 256, 0, stream>>>(Whh_f, whh16, G4 * HH);
  cvt_kernel<<<2048, 256, 0, stream>>>(Whh_b, whh16 + (size_t)G4 * HH, G4 * HH);
  cvt_kernel<<<1024, 256, 0, stream>>>(Wih_f, wih16, G4 * DIN);
  cvt_kernel<<<1024, 256, 0, stream>>>(Wih_b, wih16 + (size_t)G4 * DIN, G4 * DIN);
  cvt_kernel<<<512, 256, 0, stream>>>(Wlin, wlin16, DOUT * H2);
  bias_kernel<<<16, 256, 0, stream>>>(bih_f, bhh_f, biasc, G4);
  bias_kernel<<<16, 256, 0, stream>>>(bih_b, bhh_b, biasc + G4, G4);
  zero_kernel<<<256, 256, 0, stream>>>(hbuf, 4 * BB * HH);
  zbar_kernel<<<1, 32, 0, stream>>>(bar);

  lstm_kernel<<<2 * NWG_DIR, 256, 0, stream>>>(xf16, whh16, wih16, biasc,
                                               hbuf, bar);
  linear_kernel<<<16, 256, 0, stream>>>(hbuf, wlin16, blin, out);
}
